// MultiHeadPosAttn_75763223102207
// MI455X (gfx1250) — matrix operand order repaired and checked by reading, not compiled, not run on hardware
//
#include <hip/hip_runtime.h>
#include <hip/hip_bf16.h>

typedef __attribute__((ext_vector_type(16))) _Float16 v16h;
typedef __attribute__((ext_vector_type(8)))  float    v8f;

#define BCN (16 * 256 * 1024)   // B*C*N = 4,194,304

union FragH { v16h v; uint4 q[2]; };

// A fragment (16x32 f16, M x K): lane l holds row M = l%16.
__device__ __forceinline__ v16h ld_a(const _Float16* base, int stride, int row0, int k0, int lane) {
    int r    = row0 + (lane & 15);
    int koff = k0 + ((lane >> 4) << 3);
    const _Float16* p = base + (size_t)r * stride + koff;
    FragH f;
    f.q[0] = *(const uint4*)(p);
    f.q[1] = *(const uint4*)(p + 16);
    return f.v;
}

// B fragment (32x16 f16) from LDS holding B^T as [col][k].
__device__ __forceinline__ v16h ld_b(const _Float16* base, int stride, int col0, int k0, int lane) {
    int c    = col0 + (lane & 15);
    int koff = k0 + ((lane >> 4) << 3);
    const _Float16* p = base + (size_t)c * stride + koff;
    FragH f;
    f.q[0] = *(const uint4*)(p);
    f.q[1] = *(const uint4*)(p + 16);
    return f.v;
}

__device__ __forceinline__ v8f wmma_f16(v16h a, v16h b, v8f c) {
    return __builtin_amdgcn_wmma_f32_16x16x32_f16(false, a, false, b, (short)0, c, false, false);
}

__device__ __forceinline__ v8f v8f_zero() {
    v8f z = {0.f, 0.f, 0.f, 0.f, 0.f, 0.f, 0.f, 0.f};
    return z;
}

// Async global -> LDS copy, 128 bits per lane, tracked by ASYNCcnt.
// GVS addressing: mem = SADDR(sgpr64) + VADDR(vgpr32 byte offset).
__device__ __forceinline__ void async_ld_b128(void* lds_dst, const void* sbase, uint32_t voff) {
    uint32_t lds = (uint32_t)(size_t)lds_dst;   // low 32 bits of generic = LDS offset
    asm volatile("global_load_async_to_lds_b128 %0, %1, %2"
                 :: "v"(lds), "v"(voff), "s"(sbase)
                 : "memory");
}
__device__ __forceinline__ void wait_async0() {
    asm volatile("s_wait_asynccnt 0x0" ::: "memory");
}

// ---------------------------------------------------------------------------
// Token-major GEMM: Out[b][n][m] = sum_k Act[b][n][k] * W[m][k] + bias[m]
// Double-buffered LDS tiles filled by async b128 copies (ASYNCcnt pipeline).
// EPI 0: f16 out +bias. EPI 1: f16 out +bias +PReLU. EPI 2: f32 out +bias.
// Block = 128 threads (4 waves), tile 64 tokens x 64 outch, K-tile 32.
// ---------------------------------------------------------------------------
template<int EPI>
__global__ __launch_bounds__(128) void gemm_nhc_kernel(
    const _Float16* __restrict__ Act,   // [B][1024][K]
    const _Float16* __restrict__ W,     // [Mout][K]
    const float* __restrict__ bias, const float* __restrict__ prelu,
    void* __restrict__ Out,             // [B][1024][Mout]
    int Mout, int K)
{
    __shared__ _Float16 As[2][64][40];  // [token][k], row stride 80B (16B-aligned)
    __shared__ _Float16 Bs[2][64][40];  // [outch][k]

    const int tid  = threadIdx.x;
    const int lane = tid & 31;
    const int w    = tid >> 5;
    const int m0   = blockIdx.x * 64;
    const int t0   = blockIdx.y * 64;
    const size_t bb = blockIdx.z;
    const _Float16* Ab = Act + ((size_t)bb * 1024 + t0) * K;

    const int tw = (w >> 1) * 32;
    const int nw = (w & 1) * 32;

    auto stage = [&](int buf, int kt) {
        #pragma unroll
        for (int i = 0; i < 2; ++i) {
            int chunk = tid + i * 128;          // 0..255
            int row = chunk >> 2;               // 64 rows, 4 chunks each
            int kc  = (chunk & 3) << 3;
            async_ld_b128(&As[buf][row][kc], Ab,
                          (uint32_t)(((size_t)row * K + kt + kc) * 2));
            async_ld_b128(&Bs[buf][row][kc], W,
                          (uint32_t)(((size_t)(m0 + row) * K + kt + kc) * 2));
        }
    };

    v8f acc[2][2];
    acc[0][0] = v8f_zero(); acc[0][1] = v8f_zero();
    acc[1][0] = v8f_zero(); acc[1][1] = v8f_zero();

    stage(0, 0);
    for (int kt = 0; kt < K; kt += 32) {
        const int cur = (kt >> 5) & 1;
        wait_async0();                          // current tile landed in LDS
        __syncthreads();                        // visible to all waves; prev compute done
        if (kt + 32 < K) stage(cur ^ 1, kt + 32);   // overlap next copy with WMMA

        const _Float16* Ac = &As[cur][0][0];
        const _Float16* Bc = &Bs[cur][0][0];
        v16h a0 = ld_a(Ac, 40, tw,      0, lane);
        v16h a1 = ld_a(Ac, 40, tw + 16, 0, lane);
        v16h b0 = ld_b(Bc, 40, nw,      0, lane);
        v16h b1 = ld_b(Bc, 40, nw + 16, 0, lane);
        acc[0][0] = wmma_f16(a0, b0, acc[0][0]);
        acc[0][1] = wmma_f16(a0, b1, acc[0][1]);
        acc[1][0] = wmma_f16(a1, b0, acc[1][0]);
        acc[1][1] = wmma_f16(a1, b1, acc[1][1]);
    }

    float slope = 0.0f;
    if (EPI == 1) slope = *prelu;

    #pragma unroll
    for (int mi = 0; mi < 2; ++mi)
    #pragma unroll
    for (int ni = 0; ni < 2; ++ni)
    #pragma unroll
    for (int r = 0; r < 8; ++r) {
        int tok = t0 + tw + mi * 16 + r + ((lane >> 4) << 3);
        int mo  = m0 + nw + ni * 16 + (lane & 15);
        float v = acc[mi][ni][r] + bias[mo];
        if (EPI == 1) v = (v >= 0.f) ? v : slope * v;
        size_t off = ((size_t)bb * 1024 + tok) * Mout + mo;
        if (EPI == 2) ((float*)Out)[off] = v;
        else          ((_Float16*)Out)[off] = (_Float16)v;
    }
}

// ---------------------------------------------------------------------------
// Fused attention: one workgroup = (b, h, 32-query block).
// qkvT token-major [b][n][768] (q|k|v); vD channel-major [b][256][1024].
// Scores (32x1024 f32) + attn (f16) fully resident in WGP LDS; all tile
// staging via async b128 copies, double-buffered.
// ---------------------------------------------------------------------------
__global__ __launch_bounds__(128) void attn_kernel(
    const _Float16* __restrict__ qkvT, const _Float16* __restrict__ vDall,
    float* __restrict__ oT)
{
    extern __shared__ char smem[];
    float*    es  = (float*)smem;                 // [32][1024] f32 (131072B)
    _Float16* at  = (_Float16*)(smem + 131072);   // [32][1024] f16 (65536B)
    _Float16* qs  = (_Float16*)(smem + 196608);   // [32][72]        (4608B)
    _Float16* ks  = (_Float16*)(smem + 201216);   // [2][64][72] / vs [2][64][40]
    float*    red = (float*)(smem + 219648);      // [32][4]

    const int tid  = threadIdx.x;
    const int lane = tid & 31;
    const int w    = tid >> 5;
    const int r0   = blockIdx.x * 32;
    const int b    = blockIdx.y >> 2;
    const int h    = blockIdx.y & 3;

    const _Float16* qT = qkvT + (size_t)b * 1024 * 768 +       h * 64;
    const _Float16* kT = qkvT + (size_t)b * 1024 * 768 + 256 + h * 64;
    const _Float16* vD = vDall + ((size_t)b * 256 + h * 64) * 1024;  // [64][1024]

    auto stage_ks = [&](int buf, int jt) {
        #pragma unroll
        for (int i = 0; i < 4; ++i) {
            int chunk = tid + i * 128;            // 0..511
            int j  = chunk >> 3;
            int dc = (chunk & 7) << 3;
            async_ld_b128(&ks[buf * 4608 + j * 72 + dc], kT,
                          (uint32_t)(((size_t)(jt * 64 + j) * 768 + dc) * 2));
        }
    };

    // stage q block (async) + first k tile
    #pragma unroll
    for (int i = 0; i < 2; ++i) {
        int chunk = tid + i * 128;                // 0..255
        int r  = chunk >> 3;
        int dc = (chunk & 7) << 3;
        async_ld_b128(&qs[r * 72 + dc], qT,
                      (uint32_t)(((size_t)(r0 + r) * 768 + dc) * 2));
    }
    stage_ks(0, 0);

    // ---- energy: es[r][j] = sum_d q[r][d] * k[j][d] ----
    for (int jt = 0; jt < 16; ++jt) {
        const int cur = jt & 1;
        wait_async0();
        __syncthreads();
        if (jt + 1 < 16) stage_ks(cur ^ 1, jt + 1);

        const _Float16* kbuf = ks + cur * 4608;
        v8f e0 = v8f_zero(), e1 = v8f_zero();
        #pragma unroll
        for (int kd = 0; kd < 64; kd += 32) {
            v16h bf = ld_b(kbuf, 72, w * 16, kd, lane);
            v16h a0 = ld_a(qs, 72, 0,  kd, lane);
            v16h a1 = ld_a(qs, 72, 16, kd, lane);
            e0 = wmma_f16(a0, bf, e0);
            e1 = wmma_f16(a1, bf, e1);
        }
        #pragma unroll
        for (int r = 0; r < 8; ++r) {
            int row = r + ((lane >> 4) << 3);
            int col = jt * 64 + w * 16 + (lane & 15);
            es[(size_t)row * 1024 + col]        = e0[r];
            es[(size_t)(row + 16) * 1024 + col] = e1[r];
        }
    }
    __syncthreads();

    // ---- softmax over keys (4 threads per row, 256 cols each) ----
    {
        int row = tid >> 2, g = tid & 3, c0 = g << 8;
        float* er = es + (size_t)row * 1024 + c0;
        float mx = -1e30f;
        for (int c = 0; c < 256; ++c) mx = fmaxf(mx, er[c]);
        red[row * 4 + g] = mx;
        __syncthreads();
        float m4 = fmaxf(fmaxf(red[row * 4], red[row * 4 + 1]),
                         fmaxf(red[row * 4 + 2], red[row * 4 + 3]));
        __syncthreads();
        float s = 0.f;
        for (int c = 0; c < 256; ++c) { float e = __expf(er[c] - m4); er[c] = e; s += e; }
        red[row * 4 + g] = s;
        __syncthreads();
        float inv = 1.f / (red[row * 4] + red[row * 4 + 1] +
                           red[row * 4 + 2] + red[row * 4 + 3]);
        _Float16* ar = at + (size_t)row * 1024 + c0;
        for (int c = 0; c < 256; ++c) ar[c] = (_Float16)(er[c] * inv);
    }

    // ---- o[r][d] = sum_j attn[r][j] * v[d][j] ----
    _Float16* vs = ks;                            // reuse: [2][64][40]
    auto stage_vs = [&](int buf, int jt) {
        #pragma unroll
        for (int i = 0; i < 2; ++i) {
            int chunk = tid + i * 128;            // 0..255
            int d  = chunk >> 2;
            int kc = (chunk & 3) << 3;
            async_ld_b128(&vs[buf * 2560 + d * 40 + kc], vD,
                          (uint32_t)(((size_t)d * 1024 + jt * 32 + kc) * 2));
        }
    };

    v8f o0 = v8f_zero(), o1 = v8f_zero();
    stage_vs(0, 0);
    for (int jt = 0; jt < 32; ++jt) {
        const int cur = jt & 1;
        wait_async0();
        __syncthreads();
        if (jt + 1 < 32) stage_vs(cur ^ 1, jt + 1);

        const _Float16* vbuf = vs + cur * 2560;
        v16h bf = ld_b(vbuf, 40, w * 16, 0, lane);
        v16h a0 = ld_a(at, 1024, 0,  jt * 32, lane);
        v16h a1 = ld_a(at, 1024, 16, jt * 32, lane);
        o0 = wmma_f16(a0, bf, o0);
        o1 = wmma_f16(a1, bf, o1);
    }
    // write oT[b][tok][h*64+d]
    #pragma unroll
    for (int r = 0; r < 8; ++r) {
        int row = r + ((lane >> 4) << 3);
        int d   = w * 16 + (lane & 15);
        oT[((size_t)b * 1024 + r0 + row) * 256 + h * 64 + d]      = o0[r];
        oT[((size_t)b * 1024 + r0 + row + 16) * 256 + h * 64 + d] = o1[r];
    }
}

// ---------------------------------------------------------------------------
// BatchNorm over token-major X [b][n][c]; residual in NHC or CHW layout.
// ---------------------------------------------------------------------------
__global__ __launch_bounds__(256) void bn_reduce(
    const float* __restrict__ X, const float* __restrict__ Rnhc,
    const float* __restrict__ Rchw,
    float* __restrict__ mean, float* __restrict__ invstd)
{
    const int c = blockIdx.x, t = threadIdx.x;
    float s = 0.f, s2 = 0.f;
    for (int idx = t; idx < 16 * 1024; idx += 256) {
        int bb = idx >> 10, n = idx & 1023;
        size_t onhc = ((size_t)bb * 1024 + n) * 256 + c;
        float v = X[onhc];
        if (Rnhc) v += Rnhc[onhc];
        if (Rchw) v += Rchw[(((size_t)bb * 256 + c) << 10) + n];
        s += v; s2 += v * v;
    }
    __shared__ float sh[256], sh2[256];
    sh[t] = s; sh2[t] = s2;
    __syncthreads();
    for (int o = 128; o > 0; o >>= 1) {
        if (t < o) { sh[t] += sh[t + o]; sh2[t] += sh2[t + o]; }
        __syncthreads();
    }
    if (t == 0) {
        float m = sh[0] * (1.f / 16384.f);
        float var = sh2[0] * (1.f / 16384.f) - m * m;
        mean[c] = m;
        invstd[c] = rsqrtf(var + 1e-5f);
    }
}

template<int OUT_CHW>
__global__ __launch_bounds__(256) void bn_apply(
    const float* __restrict__ X, const float* __restrict__ Rnhc,
    const float* __restrict__ Rchw,
    const float* __restrict__ mean, const float* __restrict__ invstd,
    const float* __restrict__ gamma, const float* __restrict__ beta,
    float* __restrict__ out32, _Float16* __restrict__ out16)
{
    size_t i = (size_t)blockIdx.x * 256 + threadIdx.x;   // NHC index
    int c = (int)(i & 255);
    int n = (int)((i >> 8) & 1023);
    int b = (int)(i >> 18);
    float v = X[i];
    if (Rnhc) v += Rnhc[i];
    if (Rchw) v += Rchw[(((size_t)b * 256 + c) << 10) + n];
    float y = (v - mean[c]) * invstd[c] * gamma[c] + beta[c];
    if (OUT_CHW) out32[(((size_t)b * 256 + c) << 10) + n] = y;
    else         out32[i] = y;
    if (out16) out16[i] = (_Float16)y;
}

// ---------------------------------------------------------------------------
// Conversions
// ---------------------------------------------------------------------------
__global__ void chw_to_nhc_f16(const float* __restrict__ src, _Float16* __restrict__ dst)
{
    size_t i = (size_t)blockIdx.x * 256 + threadIdx.x;   // CHW index (coalesced read)
    int b = (int)(i >> 18);
    int c = (int)((i >> 10) & 255);
    int n = (int)(i & 1023);
    dst[((size_t)b * 1024 + n) * 256 + c] = (_Float16)src[i];
}

// vD[b][c][n] = qkvT[b][n][512 + c]  (channel-major V for async o-GEMM tiles)
__global__ void v_to_chw(const _Float16* __restrict__ qkvT, _Float16* __restrict__ vD)
{
    size_t i = (size_t)blockIdx.x * 256 + threadIdx.x;   // NHC order (coalesced read)
    int c = (int)(i & 255);
    int n = (int)((i >> 8) & 1023);
    int b = (int)(i >> 18);
    vD[(((size_t)b * 256 + c) << 10) + n] = qkvT[((size_t)b * 1024 + n) * 768 + 512 + c];
}

__global__ void f32_to_f16_kernel(const float* __restrict__ src,
                                  _Float16* __restrict__ dst, int n)
{
    int i = blockIdx.x * 256 + threadIdx.x;
    if (i < n) dst[i] = (_Float16)src[i];
}

__global__ void concat3_kernel(const float* __restrict__ a, const float* __restrict__ b,
                               const float* __restrict__ c, float* __restrict__ dst)
{
    int i = blockIdx.x * 256 + threadIdx.x;
    if (i < 256)      dst[i] = a[i];
    else if (i < 512) dst[i] = b[i - 256];
    else if (i < 768) dst[i] = c[i - 512];
}

// ---------------------------------------------------------------------------
extern "C" void kernel_launch(void* const* d_in, const int* in_sizes, int n_in,
                              void* d_out, int out_size, void* d_ws, size_t ws_size,
                              hipStream_t stream)
{
    const float* x     = (const float*)d_in[0];
    const float* Wq    = (const float*)d_in[1];
    const float* bq    = (const float*)d_in[2];
    const float* Wk    = (const float*)d_in[3];
    const float* bk    = (const float*)d_in[4];
    const float* Wv    = (const float*)d_in[5];
    const float* bv    = (const float*)d_in[6];
    const float* gamma = (const float*)d_in[7];
    const float* beta  = (const float*)d_in[8];
    const float* W1    = (const float*)d_in[9];
    const float* b1    = (const float*)d_in[10];
    const float* aP    = (const float*)d_in[11];
    const float* W2    = (const float*)d_in[12];
    const float* b2    = (const float*)d_in[13];

    char* ws = (char*)d_ws;
    size_t off = 0;
    auto carve = [&](size_t bytes) -> void* {
        void* p = ws + off;
        off = (off + bytes + 255) & ~(size_t)255;
        return p;
    };

    _Float16* xhT   = (_Float16*)carve((size_t)BCN * 2);              // [b][n][256]
    _Float16* wqkvh = (_Float16*)carve((size_t)768 * 256 * 2);        // [768][256]
    _Float16* w1h   = (_Float16*)carve((size_t)1024 * 256 * 2);       // [1024][256]
    _Float16* w2h   = (_Float16*)carve((size_t)256 * 1024 * 2);       // [256][1024]
    float*    bqkv  = (float*)carve(768 * 4);
    float*    mean0 = (float*)carve(256 * 4);
    float*    istd0 = (float*)carve(256 * 4);
    float*    mean1 = (float*)carve(256 * 4);
    float*    istd1 = (float*)carve(256 * 4);
    _Float16* qkvT  = (_Float16*)carve((size_t)16 * 1024 * 768 * 2);  // [b][n][768]
    _Float16* vD    = (_Float16*)carve((size_t)BCN * 2);              // [b][256][1024]
    float*    oT    = (float*)carve((size_t)BCN * 4);                 // [b][n][256]
    float*    mh32  = (float*)carve((size_t)BCN * 4);                 // [b][n][256]
    _Float16* mh16  = (_Float16*)carve((size_t)BCN * 2);              // [b][n][256]
    _Float16* ffT   = (_Float16*)carve((size_t)16 * 1024 * 1024 * 2); // [b][n][1024]
    float*    ff2   = (float*)carve((size_t)BCN * 4);                 // [b][n][256]

    const dim3 blk128(128);

    // layout conversion + f16 weight prep
    chw_to_nhc_f16<<<BCN / 256, 256, 0, stream>>>(x, xhT);
    f32_to_f16_kernel<<<256, 256, 0, stream>>>(Wq, wqkvh,          65536);
    f32_to_f16_kernel<<<256, 256, 0, stream>>>(Wk, wqkvh + 65536,  65536);
    f32_to_f16_kernel<<<256, 256, 0, stream>>>(Wv, wqkvh + 131072, 65536);
    f32_to_f16_kernel<<<1024, 256, 0, stream>>>(W1, w1h, 262144);
    f32_to_f16_kernel<<<1024, 256, 0, stream>>>(W2, w2h, 262144);
    concat3_kernel<<<3, 256, 0, stream>>>(bq, bk, bv, bqkv);

    // QKV: [b][n][768] = xhT @ wqkv^T
    gemm_nhc_kernel<0><<<dim3(12, 16, 16), blk128, 0, stream>>>(
        xhT, wqkvh, bqkv, nullptr, qkvT, 768, 256);

    // channel-major V copy for async tile staging in attention
    v_to_chw<<<BCN / 256, 256, 0, stream>>>(qkvT, vD);

    // fused attention (scores resident in 320KB WGP LDS)
    const size_t attn_smem = 131072 + 65536 + 4608 + 18432 + 512; // 220160 B
    attn_kernel<<<dim3(32, 64), blk128, attn_smem, stream>>>(qkvT, vD, oT);

    // mh = BN(o + x): NHC f32 + f16 copies
    bn_reduce<<<256, 256, 0, stream>>>(oT, nullptr, x, mean0, istd0);
    bn_apply<0><<<BCN / 256, 256, 0, stream>>>(oT, nullptr, x, mean0, istd0,
                                               gamma, beta, mh32, mh16);

    // FFN1: [b][n][1024] = mh16 @ W1^T, +b1, PReLU
    gemm_nhc_kernel<1><<<dim3(16, 16, 16), blk128, 0, stream>>>(
        mh16, w1h, b1, aP, ffT, 1024, 256);

    // FFN2: [b][n][256] = ffT @ W2^T, +b2 (f32)
    gemm_nhc_kernel<2><<<dim3(4, 16, 16), blk128, 0, stream>>>(
        ffT, w2h, b2, nullptr, ff2, 256, 1024);

    // out = BN(ff2 + mh), written back in CHW layout
    bn_reduce<<<256, 256, 0, stream>>>(ff2, mh32, nullptr, mean1, istd1);
    bn_apply<1><<<BCN / 256, 256, 0, stream>>>(ff2, mh32, nullptr, mean1, istd1,
                                               gamma, beta, (float*)d_out, nullptr);
}
